// Block_89498528514210
// MI455X (gfx1250) — matrix operand order repaired and checked by reading, not compiled, not run on hardware
//
#include <hip/hip_runtime.h>
#include <hip/hip_bf16.h>

// ---------------------------------------------------------------------------
// ViT encoder block for MI455X (gfx1250): bf16 WMMA GEMMs + flash attention,
// with Tensor Data Mover (TDM) LDS staging for the GEMM tiles.
// B=16, N=1024, D=768, H=12, HD=64, HID=3072.
// ---------------------------------------------------------------------------

typedef __bf16 bf16;
typedef __attribute__((ext_vector_type(16))) __bf16 bf16x16;
typedef __attribute__((ext_vector_type(8)))  float  f32x8;

struct U16B { uint4 lo, hi; };

__device__ __forceinline__ uint4 ld16(const bf16* p) {
    return *reinterpret_cast<const uint4*>(p);
}
__device__ __forceinline__ void st16(bf16* p, uint4 v) {
    *reinterpret_cast<uint4*>(p) = v;
}
__device__ __forceinline__ bf16x16 mkfrag(uint4 lo, uint4 hi) {
    U16B u; u.lo = lo; u.hi = hi;
    return __builtin_bit_cast(bf16x16, u);
}
__device__ __forceinline__ f32x8 wmma_bf16(bf16x16 a, bf16x16 b, f32x8 c) {
    // D = A(16x32) * B(32x16) + C, f32 accumulate
    return __builtin_amdgcn_wmma_f32_16x16x32_bf16(
        false, a, false, b, (short)0, c, false, false);
}

#define LDT 40  // 32 + 8 halfs padding: 16B-aligned rows, spreads banks;
                // matches TDM pad: 4 DWORD pad after every 16 DWORDs.

// A-fragment (16x32 bf16): lane holds row (lane&15); runs K=[8h,8h+8) and [16+8h,..)
__device__ __forceinline__ bf16x16 load_afrag(const bf16* tile, int rowBase, int lane) {
    const int h = lane >> 4, mr = lane & 15;
    const bf16* pr = tile + (rowBase + mr) * LDT;
    return mkfrag(ld16(pr + h * 8), ld16(pr + 16 + h * 8));
}
// B-fragment (32x16 bf16): lane holds column (lane&15); runs K=[8h,8h+8) and [16+8h,..)
__device__ __forceinline__ bf16x16 load_bfrag(const bf16* tile, int nBase, int lane) {
    const int h = lane >> 4, nc = lane & 15;
    const bf16* pr = tile + (nBase + nc) * LDT + h * 8;
    return mkfrag(ld16(pr), ld16(pr + 16));
}

// ---------------------------------------------------------------------------
// Tensor Data Mover staging (gfx1250). Descriptor per CDNA5 ISA ch.8 (D#).
// ---------------------------------------------------------------------------
#if defined(__has_builtin)
#if __has_builtin(__builtin_amdgcn_tensor_load_to_lds) && \
    __has_builtin(__builtin_amdgcn_s_wait_tensorcnt)
#define USE_TDM 1
#endif
#endif
#ifndef USE_TDM
#define USE_TDM 0
#endif

#if USE_TDM
typedef unsigned int u32x4v __attribute__((ext_vector_type(4)));
typedef int          i32x4v __attribute__((ext_vector_type(4)));
typedef int          i32x8v __attribute__((ext_vector_type(8)));

// DMA a [rows x 32] bf16 tile (row stride = strideElems bf16) from global into
// LDS at byte offset ldsAddr, padded to LDT=40-half rows (pad 4 DW / 16 DW).
__device__ __forceinline__ void tdm_load_tile(const bf16* gtile, unsigned ldsAddr,
                                              unsigned strideElems, unsigned tensorRows,
                                              unsigned rows) {
    const unsigned long long ga = (unsigned long long)(uintptr_t)gtile;
    u32x4v g0;
    g0.x = 1u;                                        // count=1 (valid user D#)
    g0.y = ldsAddr;                                   // lds_addr [63:32]
    g0.z = (unsigned)(ga & 0xFFFFFFFFu);              // global_addr[31:0]
    g0.w = (unsigned)((ga >> 32) & 0x01FFFFFFull)     // global_addr[56:32]
         | (2u << 30);                                // type=2 ("image")
    i32x8v g1;
    // data_size=1 (2B), pad_enable=1, pad_interval=3 (16 DW), pad_amount=3 (4 DW)
    g1[0] = (1 << 16) | (1 << 20) | (3 << 22) | (3 << 25);
    const unsigned td0 = strideElems;                 // tensor_dim0 (elems)
    const unsigned td1 = tensorRows;                  // tensor_dim1 (rows)
    g1[1] = (int)((td0 & 0xFFFFu) << 16);             // [31:16]=tensor_dim0[15:0]
    g1[2] = (int)(((td0 >> 16) & 0xFFFFu) | ((td1 & 0xFFFFu) << 16));
    g1[3] = (int)(((td1 >> 16) & 0xFFFFu) | (32u << 16));   // tile_dim0=32 elems
    g1[4] = (int)rows;                                // tile_dim1=rows, tile_dim2=0
    g1[5] = (int)strideElems;                         // tensor_dim0_stride[31:0]
    g1[6] = 0;                                        // stride[47:32]=0, dim1_stride lo=0
    g1[7] = 0;
    const i32x4v gz = {0, 0, 0, 0};
#if __clang_major__ >= 23
    const i32x8v gz8 = {0, 0, 0, 0, 0, 0, 0, 0};
    __builtin_amdgcn_tensor_load_to_lds(g0, g1, gz, gz, gz8, 0);
#else
    __builtin_amdgcn_tensor_load_to_lds(g0, g1, gz, gz, 0);
#endif
}
__device__ __forceinline__ unsigned lds_offset(const void* p) {
    return (unsigned)(uintptr_t)p;   // generic LDS addr: low 32 bits = LDS offset
}
#endif  // USE_TDM

// ---------------------------------------------------------------------------
// Elementwise f32 -> bf16 cast (weights)
// ---------------------------------------------------------------------------
__global__ void cast_bf16_kernel(const float* __restrict__ in, bf16* __restrict__ out, int n) {
    int i = blockIdx.x * 256 + threadIdx.x;
    if (i < n) out[i] = (bf16)in[i];
}

// ---------------------------------------------------------------------------
// LayerNorm over D=768, one block (256 thr) per row, bf16 output
// ---------------------------------------------------------------------------
__global__ __launch_bounds__(256)
void ln_bf16_kernel(const float* __restrict__ x, const float* __restrict__ g,
                    const float* __restrict__ bta, bf16* __restrict__ out) {
    const int row = blockIdx.x;
    const float* xr = x + (size_t)row * 768;
    float v[3], s = 0.f, ss = 0.f;
#pragma unroll
    for (int i = 0; i < 3; ++i) {
        v[i] = xr[threadIdx.x + i * 256];
        s += v[i]; ss += v[i] * v[i];
    }
#pragma unroll
    for (int off = 16; off >= 1; off >>= 1) {
        s  += __shfl_xor(s, off);
        ss += __shfl_xor(ss, off);
    }
    __shared__ float red[2][8];
    const int wid = threadIdx.x >> 5, lane = threadIdx.x & 31;
    if (lane == 0) { red[0][wid] = s; red[1][wid] = ss; }
    __syncthreads();
    s = 0.f; ss = 0.f;
#pragma unroll
    for (int w = 0; w < 8; ++w) { s += red[0][w]; ss += red[1][w]; }
    const float mu  = s * (1.f / 768.f);
    const float var = ss * (1.f / 768.f) - mu * mu;
    const float inv = rsqrtf(var + 1e-6f);
    bf16* o = out + (size_t)row * 768;
#pragma unroll
    for (int i = 0; i < 3; ++i) {
        const int c = threadIdx.x + i * 256;
        o[c] = (bf16)((v[i] - mu) * inv * g[c] + bta[c]);
    }
}

// ---------------------------------------------------------------------------
// Tiled bf16 WMMA GEMM:  C[M,Nn] = A[M,K] @ W[Nn,K]^T  (+ epilogue)
// BM=BN=128, BK=32; 256 threads = 8 waves (4m x 2n); each wave 32x64 (2x4 WMMA)
// Tile staging via TDM (tensor_load_to_lds) when available.
// ---------------------------------------------------------------------------
enum { EPI_QKV = 0, EPI_BIAS_RES = 1, EPI_BIAS_GELU = 2 };

template <int EPI>
__global__ __launch_bounds__(256)
void gemm_bf16_kernel(const bf16* __restrict__ A, const bf16* __restrict__ W,
                      int Nn, int K,
                      const float* __restrict__ bias,
                      const float* resid, float* outF, bf16* __restrict__ outB,
                      bf16* __restrict__ oq, bf16* __restrict__ okk, bf16* __restrict__ ovT) {
    __shared__ bf16 smA[2][128 * LDT];
    __shared__ bf16 smB[2][128 * LDT];

    const int tid  = threadIdx.x;
    const int lane = tid & 31, wid = tid >> 5;
    const int wm = wid & 3, wn = wid >> 2;
    const int m0 = blockIdx.y * 128, n0 = blockIdx.x * 128;

    f32x8 acc[2][4];
    const f32x8 zero = {0.f, 0.f, 0.f, 0.f, 0.f, 0.f, 0.f, 0.f};
#pragma unroll
    for (int i = 0; i < 2; ++i)
#pragma unroll
        for (int j = 0; j < 4; ++j) acc[i][j] = zero;

    const int KT = K >> 5;
    const bf16* Atile = A + (size_t)m0 * K;
    const bf16* Wtile = W + (size_t)n0 * K;

#if USE_TDM
    // ---- TDM staging: wave 0 DMAs both 128x32 tiles per k-step ----
    const unsigned ldsA[2] = { lds_offset(&smA[0][0]), lds_offset(&smA[1][0]) };
    const unsigned ldsB[2] = { lds_offset(&smB[0][0]), lds_offset(&smB[1][0]) };
    if (wid == 0) {
        tdm_load_tile(Atile, ldsA[0], (unsigned)K, 16384u, 128u);
        tdm_load_tile(Wtile, ldsB[0], (unsigned)K, (unsigned)Nn, 128u);
        __builtin_amdgcn_s_wait_tensorcnt(0);
    }
    __syncthreads();

    for (int kt = 0; kt < KT; ++kt) {
        const int cur = kt & 1;
        if (wid == 0 && kt + 1 < KT) {
            const int ko = (kt + 1) * 32;
            tdm_load_tile(Atile + ko, ldsA[cur ^ 1], (unsigned)K, 16384u, 128u);
            tdm_load_tile(Wtile + ko, ldsB[cur ^ 1], (unsigned)K, (unsigned)Nn, 128u);
        }
        bf16x16 af[2], bfm[4];
#pragma unroll
        for (int im = 0; im < 2; ++im) af[im]  = load_afrag(smA[cur], wm * 32 + im * 16, lane);
#pragma unroll
        for (int in = 0; in < 4; ++in) bfm[in] = load_bfrag(smB[cur], wn * 64 + in * 16, lane);
#pragma unroll
        for (int im = 0; im < 2; ++im)
#pragma unroll
            for (int in = 0; in < 4; ++in)
                acc[im][in] = wmma_bf16(af[im], bfm[in], acc[im][in]);
        if (wid == 0 && kt + 1 < KT) __builtin_amdgcn_s_wait_tensorcnt(0);
        __syncthreads();
    }
#else
    // ---- fallback: manual global->VGPR->LDS double-buffered staging ----
    const int r0 = tid >> 2, s0 = tid & 3;
    const bf16* Ag0 = Atile + (size_t)r0 * K + s0 * 8;
    const bf16* Ag1 = Ag0 + (size_t)64 * K;
    const bf16* Wg0 = Wtile + (size_t)r0 * K + s0 * 8;
    const bf16* Wg1 = Wg0 + (size_t)64 * K;
    const int ldsOff0 = r0 * LDT + s0 * 8;
    const int ldsOff1 = (r0 + 64) * LDT + s0 * 8;

    uint4 ra0 = ld16(Ag0), ra1 = ld16(Ag1), rb0 = ld16(Wg0), rb1 = ld16(Wg1);
    st16(&smA[0][ldsOff0], ra0); st16(&smA[0][ldsOff1], ra1);
    st16(&smB[0][ldsOff0], rb0); st16(&smB[0][ldsOff1], rb1);
    __syncthreads();

    for (int kt = 0; kt < KT; ++kt) {
        const int cur = kt & 1;
        if (kt + 1 < KT) {
            const int ko = (kt + 1) * 32;
            ra0 = ld16(Ag0 + ko); ra1 = ld16(Ag1 + ko);
            rb0 = ld16(Wg0 + ko); rb1 = ld16(Wg1 + ko);
        }
        bf16x16 af[2], bfm[4];
#pragma unroll
        for (int im = 0; im < 2; ++im) af[im]  = load_afrag(smA[cur], wm * 32 + im * 16, lane);
#pragma unroll
        for (int in = 0; in < 4; ++in) bfm[in] = load_bfrag(smB[cur], wn * 64 + in * 16, lane);
#pragma unroll
        for (int im = 0; im < 2; ++im)
#pragma unroll
            for (int in = 0; in < 4; ++in)
                acc[im][in] = wmma_bf16(af[im], bfm[in], acc[im][in]);
        if (kt + 1 < KT) {
            const int nxt = cur ^ 1;
            st16(&smA[nxt][ldsOff0], ra0); st16(&smA[nxt][ldsOff1], ra1);
            st16(&smB[nxt][ldsOff0], rb0); st16(&smB[nxt][ldsOff1], rb1);
        }
        __syncthreads();
    }
#endif

    // epilogue: C layout row = r + 8*(lane>=16), col = lane&15
    const int hq = lane >> 4, lc = lane & 15;
#pragma unroll
    for (int im = 0; im < 2; ++im)
#pragma unroll
        for (int in = 0; in < 4; ++in) {
            f32x8 c = acc[im][in];
#pragma unroll
            for (int r = 0; r < 8; ++r) {
                const int m = m0 + wm * 32 + im * 16 + r + 8 * hq;
                const int n = n0 + wn * 64 + in * 16 + lc;
                const float v = c[r];
                if (EPI == EPI_QKV) {
                    const int which = n / 768;
                    const int h = (n % 768) >> 6;
                    const int d = n & 63;
                    const int b = m >> 10, row = m & 1023;
                    const int bh = b * 12 + h;
                    const bf16 bv = (bf16)v;
                    if (which == 0)      oq [((size_t)bh * 1024 + row) * 64 + d] = bv;
                    else if (which == 1) okk[((size_t)bh * 1024 + row) * 64 + d] = bv;
                    else                 ovT[((size_t)bh * 64 + d) * 1024 + row] = bv;  // V transposed
                } else if (EPI == EPI_BIAS_RES) {
                    const size_t idx = (size_t)m * Nn + n;
                    outF[idx] = v + bias[n] + resid[idx];
                } else { // EPI_BIAS_GELU
                    const float t = v + bias[n];
                    const float gelu = 0.5f * t * (1.f + erff(t * 0.70710678118654752f));
                    outB[(size_t)m * Nn + n] = (bf16)gelu;
                }
            }
        }
}

// ---------------------------------------------------------------------------
// Flash attention: one wave handles 16 query rows over all 1024 keys.
// Q [bh,1024,64] bf16, K [bh,1024,64] bf16, V^T [bh,64,1024] bf16.
// Output bf16 into ao [B*N, 768] (proj-GEMM A operand).
// ---------------------------------------------------------------------------
__global__ __launch_bounds__(256)
void attn_kernel(const bf16* __restrict__ q, const bf16* __restrict__ kk,
                 const bf16* __restrict__ vT, bf16* __restrict__ ao) {
    const int tid = threadIdx.x;
    const int wid = tid >> 5, lane = tid & 31;
    const int hq = lane >> 4, lc = lane & 15;

    const int blk = blockIdx.x;          // B*H*8 blocks
    const int mt  = blk & 7;             // 128-row chunk
    const int bh  = blk >> 3;
    const int b   = bh / 12, h = bh % 12;
    const int m0  = mt * 128 + wid * 16; // this wave's 16 rows

    const bf16* qb = q  + (size_t)bh * 1024 * 64;
    const bf16* kb = kk + (size_t)bh * 1024 * 64;
    const bf16* vb = vT + (size_t)bh * 64 * 1024;

    // Q fragments: two 16x32 A-frags covering HD=64
    bf16x16 qf[2];
    {
        const bf16* pr = qb + (size_t)(m0 + lc) * 64;
#pragma unroll
        for (int c = 0; c < 2; ++c)
            qf[c] = mkfrag(ld16(pr + c * 32 + hq * 8),
                           ld16(pr + c * 32 + 16 + hq * 8));
    }

    f32x8 accO[4];
    const f32x8 zero = {0.f, 0.f, 0.f, 0.f, 0.f, 0.f, 0.f, 0.f};
#pragma unroll
    for (int t = 0; t < 4; ++t) accO[t] = zero;
    float mrow[8], lrow[8];
#pragma unroll
    for (int r = 0; r < 8; ++r) { mrow[r] = -1e30f; lrow[r] = 0.f; }

    __shared__ bf16 pbuf[8][16 * LDT];   // per-wave P tile (16 x 32 bf16, padded)
    bf16* pw = &pbuf[wid][0];

    const float scale = 0.125f;          // 1/sqrt(64)

    for (int j0 = 0; j0 < 1024; j0 += 32) {
        // S = Q K^T for two 16x16 column tiles
        f32x8 s[2];
#pragma unroll
        for (int jt = 0; jt < 2; ++jt) {
            const bf16* pr = kb + (size_t)(j0 + jt * 16 + lc) * 64 + hq * 8;
            const bf16x16 b0 = mkfrag(ld16(pr),      ld16(pr + 16));
            const bf16x16 b1 = mkfrag(ld16(pr + 32), ld16(pr + 48));
            f32x8 acc = wmma_bf16(qf[0], b0, zero);
            s[jt] = wmma_bf16(qf[1], b1, acc);
        }
        // online softmax per row (row = r + 8*hq; columns live in 16-lane halves)
        float nm[8], corr[8];
#pragma unroll
        for (int r = 0; r < 8; ++r) {
            const float a0 = s[0][r] * scale;
            const float a1 = s[1][r] * scale;
            s[0][r] = a0; s[1][r] = a1;
            float mx = fmaxf(a0, a1);
#pragma unroll
            for (int off = 8; off >= 1; off >>= 1) mx = fmaxf(mx, __shfl_xor(mx, off));
            nm[r]   = fmaxf(mrow[r], mx);
            corr[r] = __expf(mrow[r] - nm[r]);
            mrow[r] = nm[r];
        }
#pragma unroll
        for (int r = 0; r < 8; ++r) {
            const float p0 = __expf(s[0][r] - nm[r]);
            const float p1 = __expf(s[1][r] - nm[r]);
            float sum = p0 + p1;
#pragma unroll
            for (int off = 8; off >= 1; off >>= 1) sum += __shfl_xor(sum, off);
            lrow[r] = lrow[r] * corr[r] + sum;
            const int row = r + 8 * hq;
            pw[row * LDT + lc]      = (bf16)p0;
            pw[row * LDT + 16 + lc] = (bf16)p1;
        }
#pragma unroll
        for (int t = 0; t < 4; ++t)
#pragma unroll
            for (int r = 0; r < 8; ++r) accO[t][r] *= corr[r];

        // P (16x32) as A-fragment from LDS
        const bf16* ppr = &pbuf[wid][lc * LDT];
        const bf16x16 pf = mkfrag(ld16(ppr + hq * 8), ld16(ppr + 16 + hq * 8));

        // accumulate P @ V over 4 output tiles (HD = 4 x 16)
#pragma unroll
        for (int t = 0; t < 4; ++t) {
            const bf16* pr = vb + (size_t)(t * 16 + lc) * 1024 + j0 + hq * 8;
            const bf16x16 vf = mkfrag(ld16(pr), ld16(pr + 16));
            accO[t] = wmma_bf16(pf, vf, accO[t]);
        }
    }

    // finalize: O /= l, scatter to [B*N, 768]
    float inv[8];
#pragma unroll
    for (int r = 0; r < 8; ++r) inv[r] = 1.f / lrow[r];
#pragma unroll
    for (int t = 0; t < 4; ++t)
#pragma unroll
        for (int r = 0; r < 8; ++r) {
            const int row = m0 + r + 8 * hq;
            const int col = h * 64 + t * 16 + lc;
            ao[((size_t)b * 1024 + row) * 768 + col] = (bf16)(accO[t][r] * inv[r]);
        }
}

// ---------------------------------------------------------------------------
// Host orchestration
// ---------------------------------------------------------------------------
extern "C" void kernel_launch(void* const* d_in, const int* in_sizes, int n_in,
                              void* d_out, int out_size, void* d_ws, size_t ws_size,
                              hipStream_t stream) {
    (void)in_sizes; (void)n_in; (void)out_size; (void)ws_size;
    const float* x      = (const float*)d_in[0];
    const float* qkv_w  = (const float*)d_in[1];
    const float* proj_w = (const float*)d_in[2];
    const float* proj_b = (const float*)d_in[3];
    const float* fc1_w  = (const float*)d_in[4];
    const float* fc1_b  = (const float*)d_in[5];
    const float* fc2_w  = (const float*)d_in[6];
    const float* fc2_b  = (const float*)d_in[7];
    const float* n1g    = (const float*)d_in[8];
    const float* n1b    = (const float*)d_in[9];
    const float* n2g    = (const float*)d_in[10];
    const float* n2b    = (const float*)d_in[11];
    float* out = (float*)d_out;

    // workspace layout (bf16 element offsets)
    bf16* ws    = (bf16*)d_ws;
    bf16* qkvw  = ws;                         // 2304*768
    bf16* projw = qkvw  + 2304 * 768;         // 768*768
    bf16* fc1w  = projw + 768 * 768;          // 3072*768
    bf16* fc2w  = fc1w  + 3072 * 768;         // 768*3072
    bf16* xn    = fc2w  + 768 * 3072;         // 16384*768 (reused for LN2)
    bf16* qbuf  = xn    + 16384 * 768;        // [bh,1024,64]
    bf16* kbuf  = qbuf  + 16384 * 768;        // [bh,1024,64]
    bf16* vTbuf = kbuf  + 16384 * 768;        // [bh,64,1024]
    bf16* aobuf = vTbuf + 16384 * 768;        // [16384,768]
    bf16* h1buf = aobuf + 16384 * 768;        // [16384,3072]

    auto cast = [&](const float* in, bf16* o, int n) {
        cast_bf16_kernel<<<(n + 255) / 256, 256, 0, stream>>>(in, o, n);
    };
    cast(qkv_w, qkvw, 2304 * 768);
    cast(proj_w, projw, 768 * 768);
    cast(fc1_w, fc1w, 3072 * 768);
    cast(fc2_w, fc2w, 768 * 3072);

    // --- attention branch ---
    ln_bf16_kernel<<<16384, 256, 0, stream>>>(x, n1g, n1b, xn);
    gemm_bf16_kernel<EPI_QKV><<<dim3(18, 128), 256, 0, stream>>>(
        xn, qkvw, 2304, 768, nullptr, nullptr, nullptr, nullptr, qbuf, kbuf, vTbuf);
    attn_kernel<<<1536, 256, 0, stream>>>(qbuf, kbuf, vTbuf, aobuf);
    gemm_bf16_kernel<EPI_BIAS_RES><<<dim3(6, 128), 256, 0, stream>>>(
        aobuf, projw, 768, 768, proj_b, x, out, nullptr, nullptr, nullptr, nullptr);

    // --- MLP branch ---
    ln_bf16_kernel<<<16384, 256, 0, stream>>>(out, n2g, n2b, xn);
    gemm_bf16_kernel<EPI_BIAS_GELU><<<dim3(24, 128), 256, 0, stream>>>(
        xn, fc1w, 3072, 768, fc1_b, nullptr, nullptr, h1buf, nullptr, nullptr, nullptr);
    gemm_bf16_kernel<EPI_BIAS_RES><<<dim3(6, 128), 256, 0, stream>>>(
        h1buf, fc2w, 768, 3072, fc2_b, out, out, nullptr, nullptr, nullptr, nullptr);
}
